// ContentLoss_10849087389931
// MI455X (gfx1250) — matrix operand order repaired and checked by reading, not compiled, not run on hardware
//
#include <hip/hip_runtime.h>
#include <hip/hip_bf16.h>

// ContentLoss: mean |cossim(x1 gathered cols) - cossim(x2 gathered cols)|
// B=4, C=256, W=H=256 (W*H=65536), n=256 index pairs -> 1024 (b,j) pairs.
//
// Strategy (MI455X / gfx1250, wave32):
//  Kernel 1: one wave per (b, 8-pair tile, 32-channel chunk) = 1024 waves.
//    Gather f32 -> f16, pack A=[a;b] (16x32) and B=[b|a] (32x16) per the
//    CDNA5 WMMA VGPR layouts, one v_wmma_f32_16x16x32_f16 per tensor gives
//    ab/aa/bb simultaneously on diagonal bands (f32 accumulate).
//    Diagonal extraction via constant-index select tree (v_cndmask chain),
//    avoiding the LDS scratch round-trip the compiler emits for dynamic
//    vector indexing.
//  Kernel 2: single 1024-thread block combines chunk partials, computes
//    D1,D2, |D1-D2|, LDS tree-reduces to the scalar mean.

typedef __attribute__((ext_vector_type(16))) _Float16 v16h;
typedef __attribute__((ext_vector_type(8)))  float    v8f;

#define KB  4        // batch
#define KC  256      // channels
#define KWH 65536    // W*H
#define KN  256      // number of index pairs
#define KCSPLIT 8    // 8 chunks of 32 channels

__device__ __forceinline__ float select8(const v8f& v, int p)
{
    // Lane-varying index, but unrolled with constant extracts -> cndmask tree.
    float r = v[0];
    r = (p == 1) ? v[1] : r;
    r = (p == 2) ? v[2] : r;
    r = (p == 3) ? v[3] : r;
    r = (p == 4) ? v[4] : r;
    r = (p == 5) ? v[5] : r;
    r = (p == 6) ? v[6] : r;
    r = (p == 7) ? v[7] : r;
    return r;
}

__global__ __launch_bounds__(32)
void cl_pair_partial_kernel(const float* __restrict__ x1,
                            const float* __restrict__ x2,
                            const int*   __restrict__ i1,
                            const int*   __restrict__ i2,
                            float*       __restrict__ ws)
{
    const int blk    = blockIdx.x;        // 4 * 32 * 8 = 1024 blocks
    const int csplit = blk & 7;           // channel chunk
    const int ptile  = (blk >> 3) & 31;   // 8-pair tile
    const int b      = blk >> 8;          // batch
    const int lane   = threadIdx.x;       // 0..31
    const int kh     = lane >> 4;         // K-half select per ISA A/B layout
    const int mn     = lane & 15;         // row (A) / col (B) id for this lane
    const int j0     = ptile * 8;
    const int c0     = csplit * 32;

    // A row m: rows 0-7 are 'a' columns (i1), rows 8-15 are 'b' columns (i2).
    // B col n: cols 0-7 are 'b' columns (i2), cols 8-15 are 'a' columns (i1).
    const int idxA = (mn < 8) ? i1[j0 + mn] : i2[j0 + (mn - 8)];
    const int idxB = (mn < 8) ? i2[j0 + mn] : i1[j0 + (mn - 8)];

    const size_t plane = (size_t)b * KC * KWH;

    v16h a1, a2, bm1, bm2;
    // CDNA5 16-bit A 16x32 layout: lane(kh,m), VGPR r holds K = kA, kA+1 with
    //   kA = 2*(r&3) + 16*(r>>2) + 8*kh
    // CDNA5 16-bit B 32x16 layout (same K order as A):
    //   lane(kh,n), VGPR r holds K = 2*(r&3) + 16*(r>>2) + 8*kh, +1
    #pragma unroll
    for (int r = 0; r < 8; ++r) {
        const int kA = 2 * (r & 3) + 16 * (r >> 2) + 8 * kh;
        const size_t aA0 = plane + (size_t)(c0 + kA) * KWH + (size_t)idxA;
        const size_t aA1 = aA0 + KWH;
        a1[2 * r]     = (_Float16)x1[aA0];
        a1[2 * r + 1] = (_Float16)x1[aA1];
        a2[2 * r]     = (_Float16)x2[aA0];
        a2[2 * r + 1] = (_Float16)x2[aA1];

        const int kBk = 2 * (r & 3) + 16 * (r >> 2) + 8 * kh;
        const size_t aB0 = plane + (size_t)(c0 + kBk) * KWH + (size_t)idxB;
        const size_t aB1 = aB0 + KWH;
        bm1[2 * r]     = (_Float16)x1[aB0];
        bm1[2 * r + 1] = (_Float16)x1[aB1];
        bm2[2 * r]     = (_Float16)x2[aB0];
        bm2[2 * r + 1] = (_Float16)x2[aB1];
    }

    v8f acc1 = {}, acc2 = {};
    // D = A x B + C ; diagonal bands carry ab / aa / bb for 8 pairs at once.
    acc1 = __builtin_amdgcn_wmma_f32_16x16x32_f16(
        false, a1, false, bm1, (short)0, acc1, false, false);
    acc2 = __builtin_amdgcn_wmma_f32_16x16x32_f16(
        false, a2, false, bm2, (short)0, acc2, false, false);

    // C/D layout: VGPR r, lanes 0-15 -> (M=r,   N=lane),
    //                      lanes 16-31 -> (M=r+8, N=lane-16).
    // ab[p]=D[p][p]     -> VGPR p, lane p
    // aa[p]=D[p][p+8]   -> VGPR p, lane p+8
    // bb[p]=D[p+8][p]   -> VGPR p, lane p+16
    const int p = lane & 7;
    const float v1 = select8(acc1, p);
    const float v2 = select8(acc2, p);

    if (lane < 24) {
        const int q = lane >> 3;  // 0: ab, 1: aa, 2: bb
        const size_t base = (((size_t)b * 32 + ptile) * KCSPLIT + csplit) * 48;
        ws[base + (size_t)q * 8 + p]       = v1;  // x1 partials (q = 0..2)
        ws[base + (size_t)(q + 3) * 8 + p] = v2;  // x2 partials (q = 3..5)
    }
}

__global__ __launch_bounds__(1024)
void cl_reduce_kernel(const float* __restrict__ ws, float* __restrict__ out)
{
    __shared__ float red[1024];
    const int t = threadIdx.x;            // 0..1023 -> (b, j)
    const int b = t >> 8;
    const int j = t & 255;
    const int ptile = j >> 3;
    const int p = j & 7;

    float s0 = 0.f, s1 = 0.f, s2 = 0.f, s3 = 0.f, s4 = 0.f, s5 = 0.f;
    #pragma unroll
    for (int cs = 0; cs < KCSPLIT; ++cs) {
        const size_t base = (((size_t)b * 32 + ptile) * KCSPLIT + cs) * 48;
        s0 += ws[base + 0 * 8 + p];
        s1 += ws[base + 1 * 8 + p];
        s2 += ws[base + 2 * 8 + p];
        s3 += ws[base + 3 * 8 + p];
        s4 += ws[base + 4 * 8 + p];
        s5 += ws[base + 5 * 8 + p];
    }
    const float d1 = s0 / fmaxf(sqrtf(s1) * sqrtf(s2), 1e-8f);
    const float d2 = s3 / fmaxf(sqrtf(s4) * sqrtf(s5), 1e-8f);
    red[t] = fabsf(d1 - d2);
    __syncthreads();

    #pragma unroll
    for (int off = 512; off > 0; off >>= 1) {
        if (t < off) red[t] += red[t + off];
        __syncthreads();
    }
    if (t == 0) out[0] = red[0] * (1.0f / (KB * KN));
}

extern "C" void kernel_launch(void* const* d_in, const int* in_sizes, int n_in,
                              void* d_out, int out_size, void* d_ws, size_t ws_size,
                              hipStream_t stream)
{
    const float* x1 = (const float*)d_in[0];
    const float* x2 = (const float*)d_in[1];
    const int*   i1 = (const int*)d_in[2];
    const int*   i2 = (const int*)d_in[3];
    float* ws  = (float*)d_ws;   // needs 4*32*8*48 floats = 192 KB
    float* out = (float*)d_out;  // scalar f32

    // 4 batches * 32 pair-tiles * 8 channel chunks = 1024 single-wave blocks
    cl_pair_partial_kernel<<<dim3(KB * 32 * KCSPLIT), dim3(32), 0, stream>>>(
        x1, x2, i1, i2, ws);
    cl_reduce_kernel<<<dim3(1), dim3(1024), 0, stream>>>(ws, out);
    (void)in_sizes; (void)n_in; (void)out_size; (void)ws_size;
}
